// TransformerBlock_31026843746723
// MI455X (gfx1250) — matrix operand order repaired and checked by reading, not compiled, not run on hardware
//
#include <hip/hip_runtime.h>
#include <hip/hip_bf16.h>
#include <cstddef>

// ---------------------------------------------------------------------------
// Types for CDNA5 WMMA (wave32): v16bf A/B fragments, v8f accumulator.
// ---------------------------------------------------------------------------
typedef __attribute__((ext_vector_type(16))) __bf16 bf16x16;
typedef __attribute__((ext_vector_type(8)))  float  f32x8;

union Frag {
    bf16x16        v;
    unsigned int   u[8];
    unsigned short h[16];
};

static __device__ __forceinline__ f32x8 wmma_bf16(bf16x16 a, bf16x16 b, f32x8 c) {
    // D = A(16x32 bf16) * B(32x16 bf16) + C(16x16 f32)
    return __builtin_amdgcn_wmma_f32_16x16x32_bf16(
        /*neg_a=*/false, a, /*neg_b=*/false, b,
        /*c_mod=*/(short)0, c, /*reuse_a=*/false, /*reuse_b=*/false);
}

// two fp32 -> packed bf16 pair (low = a, high = b).
// gfx1250 has native v_cvt_pk_bf16_f32 (seen in compiler output); the builtin
// is not exposed on this clang and scalar casts don't fuse, so use inline asm
// to get the single packed conversion instruction.
static __device__ __forceinline__ unsigned int pack2bf(float a, float b) {
    unsigned int r;
    asm("v_cvt_pk_bf16_f32 %0, %1, %2" : "=v"(r) : "v"(a), "v"(b));
    return r;
}
static __device__ __forceinline__ unsigned short f2bf(float f) {
    return (unsigned short)pack2bf(f, f);
}

static __device__ __forceinline__ float gelu_exact(float v) {
    return 0.5f * v * (1.0f + erff(v * 0.70710678118654752f));
}

// ---------------------------------------------------------------------------
// LayerNorm: one 256-thread block per row of 768.
// ---------------------------------------------------------------------------
__global__ void ln_kernel(const float* __restrict__ x,
                          const float* __restrict__ g,
                          const float* __restrict__ bt,
                          float* __restrict__ y) {
    const int row = blockIdx.x;
    const int tid = threadIdx.x;
    const size_t base = (size_t)row * 768;

    float v[3], s = 0.f, ss = 0.f;
#pragma unroll
    for (int i = 0; i < 3; ++i) {
        v[i] = x[base + tid + i * 256];
        s  += v[i];
        ss += v[i] * v[i];
    }
    __shared__ float r1[256], r2[256];
    r1[tid] = s; r2[tid] = ss;
    __syncthreads();
    for (int st = 128; st > 0; st >>= 1) {
        if (tid < st) { r1[tid] += r1[tid + st]; r2[tid] += r2[tid + st]; }
        __syncthreads();
    }
    const float mu  = r1[0] * (1.0f / 768.0f);
    const float var = r2[0] * (1.0f / 768.0f) - mu * mu;
    const float rs  = rsqrtf(var + 1e-5f);
#pragma unroll
    for (int i = 0; i < 3; ++i) {
        const int c = tid + i * 256;
        y[base + c] = (v[i] - mu) * rs * g[c] + bt[c];
    }
}

// ---------------------------------------------------------------------------
// Tiled WMMA GEMM: out[m,n] = sum_k A[m,k] * W[n,k]  (+bias, +gelu/+residual)
// Tile: 128(M) x 64(N), K-step 64. 256 threads = 8 waves; wave grid 4x2,
// each wave owns a 32x32 block = 2x2 WMMA fragments, 8 WMMAs per K-step.
// Staging: float4 global loads, v_cvt_pk_bf16_f32 packing, ds_store_b64,
// register double-buffering so next tile's loads overlap current WMMAs.
// mode: 0 = (+bias), 1 = (+bias, gelu), 2 = (+bias, +residual)
// ---------------------------------------------------------------------------
__global__ __launch_bounds__(256)
void gemm_kernel(const float* __restrict__ A,
                 const float* __restrict__ W,
                 const float* __restrict__ bias,
                 const float* __restrict__ residual,
                 float* __restrict__ out,
                 int M, int N, int K, int mode) {
    __shared__ unsigned short lds_a[128 * 64];   // A tile, bf16 (16 KB)
    __shared__ unsigned short lds_w[64 * 64];    // W tile, bf16 (8 KB)

    const int tid  = threadIdx.x;
    const int lane = tid & 31;
    const int wv   = tid >> 5;
    const int wm   = wv & 3;          // wave M index (0..3) -> +32 rows
    const int wn   = wv >> 2;         // wave N index (0..1) -> +32 cols
    const int hi   = lane >> 4;       // half-wave
    const int ln   = lane & 15;

    const int rowBase = blockIdx.x * 128;
    const int colBase = blockIdx.y * 64;

    f32x8 acc[2][2];
#pragma unroll
    for (int a = 0; a < 2; ++a)
#pragma unroll
        for (int b = 0; b < 2; ++b) acc[a][b] = (f32x8){0,0,0,0,0,0,0,0};

    auto loadA = [&](int kt, float4* ra) {
#pragma unroll
        for (int i = 0; i < 8; ++i) {
            const int qi = i * 256 + tid;             // 2048 quads
            const int r = qi >> 4, c = (qi & 15) * 4;
            ra[i] = *(const float4*)&A[(size_t)(rowBase + r) * K + kt + c];
        }
    };
    auto loadW = [&](int kt, float4* rw) {
#pragma unroll
        for (int i = 0; i < 4; ++i) {
            const int qi = i * 256 + tid;             // 1024 quads
            const int r = qi >> 4, c = (qi & 15) * 4;
            rw[i] = *(const float4*)&W[(size_t)(colBase + r) * K + kt + c];
        }
    };

    float4 ra[8], rw[4];
    loadA(0, ra);
    loadW(0, rw);

    for (int kt = 0; kt < K; kt += 64) {
        // ---- commit staged registers -> LDS (bf16 packed, 8B stores)
#pragma unroll
        for (int i = 0; i < 8; ++i) {
            const int qi = i * 256 + tid;
            const int r = qi >> 4, c = (qi & 15) * 4;
            uint2 p; p.x = pack2bf(ra[i].x, ra[i].y); p.y = pack2bf(ra[i].z, ra[i].w);
            *(uint2*)&lds_a[r * 64 + c] = p;
        }
#pragma unroll
        for (int i = 0; i < 4; ++i) {
            const int qi = i * 256 + tid;
            const int r = qi >> 4, c = (qi & 15) * 4;
            uint2 p; p.x = pack2bf(rw[i].x, rw[i].y); p.y = pack2bf(rw[i].z, rw[i].w);
            *(uint2*)&lds_w[r * 64 + c] = p;
        }
        __syncthreads();

        // ---- issue next tile's global loads (overlap with WMMA below)
        if (kt + 64 < K) {
            loadA(kt + 64, ra);
            loadW(kt + 64, rw);
        }
        // distance-2 prefetch, uniformly guarded (no per-lane branching)
        if (kt + 128 < K) {
            const int qi = tid;
            __builtin_prefetch(&A[(size_t)(rowBase + (qi >> 4)) * K + kt + 128 + (qi & 15) * 4], 0, 0);
            __builtin_prefetch(&W[(size_t)(colBase + ((qi >> 4) & 63)) * K + kt + 128 + (qi & 15) * 4], 0, 0);
        }

        // ---- fragments + WMMA (two 32-wide K halves)
#pragma unroll
        for (int kh = 0; kh < 2; ++kh) {
            Frag fa[2], fb[2];
#pragma unroll
            for (int a = 0; a < 2; ++a) {
                const unsigned short* pa =
                    &lds_a[(wm * 32 + a * 16 + ln) * 64 + kh * 32 + hi * 8];
#pragma unroll
                for (int j = 0; j < 8; ++j)
                    fa[a].u[j] = *(const unsigned int*)(pa + ((j >> 2) * 16) + ((2 * j) & 7));
            }
#pragma unroll
            for (int b = 0; b < 2; ++b) {
                const unsigned short* pb =
                    &lds_w[(wn * 32 + b * 16 + ln) * 64 + kh * 32 + hi * 8];
#pragma unroll
                for (int j = 0; j < 8; ++j)
                    fb[b].u[j] = *(const unsigned int*)(pb + ((j >> 2) * 16) + ((2 * j) & 7));
            }
#pragma unroll
            for (int a = 0; a < 2; ++a)
#pragma unroll
                for (int b = 0; b < 2; ++b)
                    acc[a][b] = wmma_bf16(fa[a].v, fb[b].v, acc[a][b]);
        }
        __syncthreads();
    }

    // ---- epilogue (C layout: vgpr r -> row r (lanes 0-15) / r+8 (16-31))
#pragma unroll
    for (int a = 0; a < 2; ++a)
#pragma unroll
        for (int b = 0; b < 2; ++b)
#pragma unroll
            for (int r = 0; r < 8; ++r) {
                const int grow = rowBase + wm * 32 + a * 16 + hi * 8 + r;
                const int gcol = colBase + wn * 32 + b * 16 + ln;
                float v = acc[a][b][r];
                if (bias) v += bias[gcol];
                if (mode == 1) v = gelu_exact(v);
                if (mode == 2) v += residual[(size_t)grow * N + gcol];
                out[(size_t)grow * N + gcol] = v;
            }
}

// ---------------------------------------------------------------------------
// Flash attention over packed qkv [B*L, 2304] (e = qkv*768 + h*64 + d).
// One block = (b, h, 128 query rows); each of 8 waves owns 16 query rows.
// K/V tiles (32x64) staged via float4 loads (pipelined across key steps);
// V stored transposed so P*V B-fragments are contiguous pair loads.
// ---------------------------------------------------------------------------
__global__ __launch_bounds__(256)
void attn_kernel(const float* __restrict__ qkv, float* __restrict__ o) {
    constexpr int L = 1024, TD = 2304;
    __shared__ unsigned short lds_k[32 * 64];      // K tile [y][d]
    __shared__ unsigned short lds_vt[64 * 32];     // V tile transposed [d][y]
    __shared__ unsigned short lds_p[8 * 16 * 32];  // per-wave P scratch

    const int tid  = threadIdx.x;
    const int lane = tid & 31;
    const int wv   = tid >> 5;
    const int hi   = lane >> 4;
    const int ln   = lane & 15;

    const int qb = blockIdx.x & 7;          // 8 query chunks of 128
    const int bh = blockIdx.x >> 3;
    const int h  = bh % 12;
    const int b  = bh / 12;
    const int q0 = qb * 128 + wv * 16;

    // ---- load Q fragments (16x64 -> two 16x32 A-frags) via float4 runs
    Frag aq[2];
    {
        const size_t qrow = (size_t)(b * L + q0 + ln) * TD + h * 64;
#pragma unroll
        for (int kh = 0; kh < 2; ++kh)
#pragma unroll
            for (int j16 = 0; j16 < 2; ++j16) {
                const size_t d0 = qrow + kh * 32 + j16 * 16 + hi * 8;
                const float4 qa = *(const float4*)&qkv[d0];
                const float4 qc = *(const float4*)&qkv[d0 + 4];
                aq[kh].u[j16 * 4 + 0] = pack2bf(qa.x, qa.y);
                aq[kh].u[j16 * 4 + 1] = pack2bf(qa.z, qa.w);
                aq[kh].u[j16 * 4 + 2] = pack2bf(qc.x, qc.y);
                aq[kh].u[j16 * 4 + 3] = pack2bf(qc.z, qc.w);
            }
    }

    f32x8 accO[4];
#pragma unroll
    for (int nf = 0; nf < 4; ++nf) accO[nf] = (f32x8){0,0,0,0,0,0,0,0};
    float mi[8], li[8];
#pragma unroll
    for (int r = 0; r < 8; ++r) { mi[r] = -1e30f; li[r] = 0.f; }

    // staging registers (pipelined across key-step iterations)
    float4 kq[2], vq[2];
    auto loadKV = [&](int j) {
#pragma unroll
        for (int i = 0; i < 2; ++i) {
            const int qi = i * 256 + tid;              // 512 quads
            const int y = qi >> 4, d = (qi & 15) * 4;
            const size_t base = (size_t)(b * L + j + y) * TD + h * 64 + d;
            kq[i] = *(const float4*)&qkv[base + 768];
            vq[i] = *(const float4*)&qkv[base + 1536];
        }
    };
    loadKV(0);

    for (int j = 0; j < L; j += 32) {
        // ---- commit staged K/V registers -> LDS
#pragma unroll
        for (int i = 0; i < 2; ++i) {
            const int qi = i * 256 + tid;
            const int y = qi >> 4, d = (qi & 15) * 4;
            uint2 pk; pk.x = pack2bf(kq[i].x, kq[i].y); pk.y = pack2bf(kq[i].z, kq[i].w);
            *(uint2*)&lds_k[y * 64 + d] = pk;
            lds_vt[(d + 0) * 32 + y] = f2bf(vq[i].x);
            lds_vt[(d + 1) * 32 + y] = f2bf(vq[i].y);
            lds_vt[(d + 2) * 32 + y] = f2bf(vq[i].z);
            lds_vt[(d + 3) * 32 + y] = f2bf(vq[i].w);
        }
        __syncthreads();
        if (j + 32 < L) loadKV(j + 32);   // overlap next loads with compute

        // ---- S = Q * K^T  (16x32 in two 16x16 fragments)
        f32x8 s[2];
        s[0] = (f32x8){0,0,0,0,0,0,0,0};
        s[1] = (f32x8){0,0,0,0,0,0,0,0};
#pragma unroll
        for (int nf = 0; nf < 2; ++nf)
#pragma unroll
            for (int kh = 0; kh < 2; ++kh) {
                Frag bk;
                const unsigned short* p =
                    &lds_k[(nf * 16 + ln) * 64 + kh * 32 + hi * 8];
#pragma unroll
                for (int jx = 0; jx < 8; ++jx)
                    bk.u[jx] = *(const unsigned int*)(p + ((jx >> 2) * 16) + ((2 * jx) & 7));
                s[nf] = wmma_bf16(aq[kh].v, bk.v, s[nf]);
            }

        // ---- online softmax (rows live per (vgpr r, half-wave))
        float pv[2][8];
#pragma unroll
        for (int r = 0; r < 8; ++r) {
            const float s0 = s[0][r] * 0.125f;
            const float s1 = s[1][r] * 0.125f;
            float vmax = fmaxf(s0, s1);
#pragma unroll
            for (int off = 8; off >= 1; off >>= 1)
                vmax = fmaxf(vmax, __shfl_xor(vmax, off, 32));
            const float mnew  = fmaxf(mi[r], vmax);
            const float alpha = __expf(mi[r] - mnew);
            const float p0 = __expf(s0 - mnew);
            const float p1 = __expf(s1 - mnew);
            float ps = p0 + p1;
#pragma unroll
            for (int off = 8; off >= 1; off >>= 1)
                ps += __shfl_xor(ps, off, 32);
            li[r] = li[r] * alpha + ps;
            mi[r] = mnew;
#pragma unroll
            for (int nf = 0; nf < 4; ++nf) accO[nf][r] *= alpha;
            pv[0][r] = p0; pv[1][r] = p1;
        }

        // ---- P: C-layout -> LDS -> A-layout (wave-private, LDS in-order)
        unsigned short* pp = &lds_p[wv * 512];
#pragma unroll
        for (int nf = 0; nf < 2; ++nf)
#pragma unroll
            for (int r = 0; r < 8; ++r)
                pp[(hi * 8 + r) * 32 + nf * 16 + ln] = f2bf(pv[nf][r]);
        Frag fp;
        {
            const unsigned short* pa = &pp[ln * 32 + hi * 8];
#pragma unroll
            for (int jx = 0; jx < 8; ++jx)
                fp.u[jx] = *(const unsigned int*)(pa + ((jx >> 2) * 16) + ((2 * jx) & 7));
        }

        // ---- O += P * V  (four 16x16 output fragments over d=64)
#pragma unroll
        for (int nf = 0; nf < 4; ++nf) {
            Frag fvv;
            const unsigned short* pvv = &lds_vt[(nf * 16 + ln) * 32 + hi * 8];
#pragma unroll
            for (int jx = 0; jx < 8; ++jx)
                fvv.u[jx] = *(const unsigned int*)(pvv + ((jx >> 2) * 16) + ((2 * jx) & 7));
            accO[nf] = wmma_bf16(fp.v, fvv.v, accO[nf]);
        }
        __syncthreads();
    }

    // ---- finalize: O /= l, scatter back to o[B*L, 768] at head offset
#pragma unroll
    for (int nf = 0; nf < 4; ++nf)
#pragma unroll
        for (int r = 0; r < 8; ++r) {
            const int lrow = q0 + hi * 8 + r;
            o[(size_t)(b * L + lrow) * 768 + h * 64 + nf * 16 + ln] =
                accO[nf][r] / li[r];
        }
}

// ---------------------------------------------------------------------------
// Host-side orchestration.
// ---------------------------------------------------------------------------
extern "C" void kernel_launch(void* const* d_in, const int* in_sizes, int n_in,
                              void* d_out, int out_size, void* d_ws, size_t ws_size,
                              hipStream_t stream) {
    (void)in_sizes; (void)n_in; (void)out_size; (void)ws_size;
    const float* x     = (const float*)d_in[0];
    const float* ln1_g = (const float*)d_in[1];
    const float* ln1_b = (const float*)d_in[2];
    const float* w_qkv = (const float*)d_in[3];
    const float* w_out = (const float*)d_in[4];
    const float* b_out = (const float*)d_in[5];
    const float* ln2_g = (const float*)d_in[6];
    const float* ln2_b = (const float*)d_in[7];
    const float* w1    = (const float*)d_in[8];
    const float* b1    = (const float*)d_in[9];
    const float* w2    = (const float*)d_in[10];
    const float* b2    = (const float*)d_in[11];

    const int M = 8192;                 // B*L tokens
    float* ws   = (float*)d_ws;
    float* bufA = ws;                                   // 8192*3072 (qkv / h)
    float* bufB = ws + (size_t)8192 * 3072;             // 8192*768 (xn / o / xn2)
    float* bufC = bufB + (size_t)8192 * 768;            // 8192*768 (x1)
    float* out  = (float*)d_out;

    // 1) LN1
    ln_kernel<<<M, 256, 0, stream>>>(x, ln1_g, ln1_b, bufB);
    // 2) QKV = xn @ w_qkv^T   [8192, 2304]
    gemm_kernel<<<dim3(M / 128, 2304 / 64), 256, 0, stream>>>(
        bufB, w_qkv, nullptr, nullptr, bufA, M, 2304, 768, 0);
    // 3) flash attention -> o [8192, 768]
    attn_kernel<<<8 * 12 * 8, 256, 0, stream>>>(bufA, bufB);
    // 4) x1 = x + o @ w_out^T + b_out
    gemm_kernel<<<dim3(M / 128, 768 / 64), 256, 0, stream>>>(
        bufB, w_out, b_out, x, bufC, M, 768, 768, 2);
    // 5) LN2
    ln_kernel<<<M, 256, 0, stream>>>(bufC, ln2_g, ln2_b, bufB);
    // 6) h = gelu(xn2 @ w1^T + b1)  [8192, 3072]
    gemm_kernel<<<dim3(M / 128, 3072 / 64), 256, 0, stream>>>(
        bufB, w1, b1, nullptr, bufA, M, 3072, 768, 1);
    // 7) out = x1 + h @ w2^T + b2
    gemm_kernel<<<dim3(M / 128, 768 / 64), 256, 0, stream>>>(
        bufA, w2, b2, bufC, out, M, 768, 3072, 2);
}
